// Decoder_43705587204708
// MI455X (gfx1250) — matrix operand order repaired and checked by reading, not compiled, not run on hardware
//
#include <hip/hip_runtime.h>

#define HD __device__ __forceinline__

constexpr int Bb  = 256;   // batch
constexpr int Tt  = 512;   // timesteps
constexpr int Hh  = 256;   // hidden
constexpr int NWG = 16;    // persistent workgroups (each owns 16 h-columns)

typedef __bf16 bf16;
typedef __attribute__((ext_vector_type(16))) __bf16 v16bf;
typedef __attribute__((ext_vector_type(8)))  __bf16 v8bf;
typedef __attribute__((ext_vector_type(8)))  float  v8f;

HD v16bf catbf(v8bf lo, v8bf hi) {
  return __builtin_shufflevector(lo, hi, 0,1,2,3,4,5,6,7,8,9,10,11,12,13,14,15);
}

HD v8f vzero() {
  v8f v;
#pragma unroll
  for (int i = 0; i < 8; ++i) v[i] = 0.f;
  return v;
}

// float -> bf16, round-to-nearest-even (bit manipulation; avoids cast legality issues)
HD bf16 tobf(float f) {
  union { float f; unsigned u; } v; v.f = f;
  unsigned r = (v.u + 0x7FFFu + ((v.u >> 16) & 1u)) >> 16;
  union { unsigned short s; bf16 b; } o; o.s = (unsigned short)r;
  return o.b;
}

HD float sigm(float x) { return 1.f / (1.f + __expf(-x)); }

// ---- WMMA fragment loaders -------------------------------------------------
// A (16x32 bf16, MxK): lanes 0-15 hold row M=lane, K 0-7 (v0-3) and 16-23 (v4-7);
// lanes 16-31 hold row M=lane-16, K 8-15 and 24-31.  Row-major src, stride ld.
HD v16bf load_a(const bf16* src, int ld, int mrow, int kb, int lane) {
  const int row = mrow + (lane & 15);
  const int hi  = lane >> 4;                    // 0 -> K+0/K+16 ; 1 -> K+8/K+24
  const bf16* p = src + row * ld + kb + hi * 8;
  v8bf lo = *(const v8bf*)(p);
  v8bf up = *(const v8bf*)(p + 16);
  return catbf(lo, up);
}

// B (32x16 bf16, KxN): lanes 0-15 hold col N=lane, K kb+0..7 and kb+16..23;
// lanes 16-31 hold col N=lane-16, K kb+8..15 and kb+24..31.
// LDS weight slice stored column-major over K: w[n*256 + k].
HD v16bf load_b(const bf16* w, int nb, int kb, int lane) {
  const int col = nb + (lane & 15);
  const int kr  = kb + (lane >> 4) * 8;
  const bf16* p = w + col * 256 + kr;
  v8bf lo = *(const v8bf*)(p);
  v8bf up = *(const v8bf*)(p + 16);
  return catbf(lo, up);
}

HD v8f wmma_bf16(v16bf a, v16bf b, v8f c) {
  return __builtin_amdgcn_wmma_f32_16x16x32_bf16(false, a, false, b, (short)0, c,
                                                 false, false);
}

// ---- device-wide barrier over NWG co-resident workgroups -------------------
HD void gbar(unsigned* cnt, unsigned& phase) {
  __threadfence();
  __syncthreads();
  if (threadIdx.x == 0) {
    atomicAdd(cnt, 1u);
    ++phase;
    const unsigned target = phase * (unsigned)NWG;
    while (*(volatile unsigned*)cnt < target) __builtin_amdgcn_s_sleep(2);
    __threadfence();
  }
  __syncthreads();
}

// ---------------------------------------------------------------------------
__global__ void init_kernel(const float* z, const float* b_out, float* preds,
                            bf16* hbuf, unsigned* cnt, float* out) {
  const int i = blockIdx.x * blockDim.x + threadIdx.x;
  const int stride = gridDim.x * blockDim.x;
  if (i == 0) { *cnt = 0u; *out = 0.f; }
  const float b0 = b_out[0];
  for (int idx = i; idx < Tt * Bb; idx += stride) preds[idx] = b0;
  for (int idx = i; idx < Bb * Hh; idx += stride) {
    bf16 v = tobf(z[idx]);
    hbuf[idx] = v;                    // layer0 slot0 (t=0 reads slot t&1 == 0)
    hbuf[2 * Bb * Hh + idx] = v;      // layer1 slot0
  }
}

// ---------------------------------------------------------------------------
__global__ __launch_bounds__(256) void lstm_persistent(
    const float* __restrict__ z,
    const float* __restrict__ Wih0, const float* __restrict__ Whh0,
    const float* __restrict__ bih0, const float* __restrict__ bhh0,
    const float* __restrict__ Wih1, const float* __restrict__ Whh1,
    const float* __restrict__ bih1, const float* __restrict__ bhh1,
    const float* __restrict__ Wout,
    float* preds, bf16* hbuf, unsigned* cnt) {
  // LDS: three 64x256 bf16 weight stripes (col-major over K) + small tables.
  __shared__ bf16  sW0 [64 * 256];   // W_hh_l0 stripe
  __shared__ bf16  sWi1[64 * 256];   // W_ih_l1 stripe
  __shared__ bf16  sWh1[64 * 256];   // W_hh_l1 stripe
  __shared__ float sBias0[64], sBias1[64], sWih0[64], sWoutS[16];
  __shared__ float xsh[Bb];

  const int tid   = threadIdx.x;
  const int wg    = blockIdx.x;      // 0..15
  const int cblk  = wg * 16;         // this WG's h-column block
  const int lane  = tid & 31;
  const int wv    = tid >> 5;        // 0..7
  const int mbase = wv * 32;         // this wave's 32 batch rows
  const int j     = lane & 15;       // column within 16-wide tile
  const int rgrp  = (lane >> 4) << 3;// row sub-offset (0 or 8) inside a C tile

  // ---- stage weight stripes into LDS (fp32 -> bf16, gate-major columns) ----
  for (int idx = tid; idx < 64 * 256; idx += 256) {
    const int n = idx >> 8, k = idx & 255;
    const int gate = n >> 4, jj = n & 15;
    const int r = gate * Hh + cblk + jj;           // global gate row (i,f,g,o order)
    sW0 [n * 256 + k] = tobf(Whh0[r * Hh + k]);
    sWi1[n * 256 + k] = tobf(Wih1[r * Hh + k]);
    sWh1[n * 256 + k] = tobf(Whh1[r * Hh + k]);
  }
  if (tid < 64) {
    const int gate = tid >> 4, jj = tid & 15;
    const int r = gate * Hh + cblk + jj;
    sBias0[tid] = bih0[r] + bhh0[r];
    sBias1[tid] = bih1[r] + bhh1[r];
    sWih0[tid]  = Wih0[r];                          // F == 1
  }
  if (tid < 16) sWoutS[tid] = Wout[cblk + tid];
  __syncthreads();

  // ---- cell state lives in VGPRs for the whole run ----
  v8f c0[2], c1[2];
#pragma unroll
  for (int m = 0; m < 2; ++m)
#pragma unroll
    for (int r = 0; r < 8; ++r) {
      const int row = mbase + m * 16 + rgrp + r;
      const float zz = z[row * Hh + cblk + j];
      c0[m][r] = zz; c1[m][r] = zz;
    }

  unsigned phase = 0;

  for (int t = 0; t < Tt; ++t) {
    const bf16* h0p = hbuf + ((t) & 1) * (Bb * Hh);
    const bf16* h1p = hbuf + 2 * (Bb * Hh) + ((t) & 1) * (Bb * Hh);
    bf16*       h0n = hbuf + ((t + 1) & 1) * (Bb * Hh);
    bf16*       h1n = hbuf + 2 * (Bb * Hh) + ((t + 1) & 1) * (Bb * Hh);

    // autoregressive input x = pred from step t-1 (0 at t=0)
    xsh[tid] = (t == 0) ? 0.f : preds[(t - 1) * Bb + tid];
    __syncthreads();

    // ================= layer 0: gates = h0p @ Whh0.T (+ x rank-1 + bias) ====
    v8f acc[2][4];
#pragma unroll
    for (int m = 0; m < 2; ++m)
#pragma unroll
      for (int n = 0; n < 4; ++n) acc[m][n] = vzero();

    for (int ks = 0; ks < 8; ++ks) {
      const int kb = ks * 32;
      v16bf a0 = load_a(h0p, Hh, mbase,      kb, lane);
      v16bf a1 = load_a(h0p, Hh, mbase + 16, kb, lane);
#pragma unroll
      for (int nt = 0; nt < 4; ++nt) {
        v16bf b = load_b(sW0, nt * 16, kb, lane);
        acc[0][nt] = wmma_bf16(a0, b, acc[0][nt]);
        acc[1][nt] = wmma_bf16(a1, b, acc[1][nt]);
      }
    }
#pragma unroll
    for (int m = 0; m < 2; ++m) {
#pragma unroll
      for (int r = 0; r < 8; ++r) {
        const int row = mbase + m * 16 + rgrp + r;
        const float xv = xsh[row];
        const float gi = acc[m][0][r] + sBias0[j]      + xv * sWih0[j];
        const float gf = acc[m][1][r] + sBias0[16 + j] + xv * sWih0[16 + j];
        const float gz = acc[m][2][r] + sBias0[32 + j] + xv * sWih0[32 + j];
        const float go = acc[m][3][r] + sBias0[48 + j] + xv * sWih0[48 + j];
        const float cn = sigm(gf) * c0[m][r] + sigm(gi) * tanhf(gz);
        c0[m][r] = cn;
        h0n[row * Hh + cblk + j] = tobf(sigm(go) * tanhf(cn));
      }
    }
    gbar(cnt, phase);   // h0n visible device-wide

    // ============ layer 1: gates = h0n @ Wih1.T + h1p @ Whh1.T + bias =======
#pragma unroll
    for (int m = 0; m < 2; ++m)
#pragma unroll
      for (int n = 0; n < 4; ++n) acc[m][n] = vzero();

    for (int ks = 0; ks < 8; ++ks) {
      const int kb = ks * 32;
      v16bf a0 = load_a(h0n, Hh, mbase,      kb, lane);
      v16bf a1 = load_a(h0n, Hh, mbase + 16, kb, lane);
#pragma unroll
      for (int nt = 0; nt < 4; ++nt) {
        v16bf b = load_b(sWi1, nt * 16, kb, lane);
        acc[0][nt] = wmma_bf16(a0, b, acc[0][nt]);
        acc[1][nt] = wmma_bf16(a1, b, acc[1][nt]);
      }
      v16bf d0 = load_a(h1p, Hh, mbase,      kb, lane);
      v16bf d1 = load_a(h1p, Hh, mbase + 16, kb, lane);
#pragma unroll
      for (int nt = 0; nt < 4; ++nt) {
        v16bf b = load_b(sWh1, nt * 16, kb, lane);
        acc[0][nt] = wmma_bf16(d0, b, acc[0][nt]);
        acc[1][nt] = wmma_bf16(d1, b, acc[1][nt]);
      }
    }
#pragma unroll
    for (int m = 0; m < 2; ++m) {
      float prow[8];
#pragma unroll
      for (int r = 0; r < 8; ++r) {
        const int row = mbase + m * 16 + rgrp + r;
        const float gi = acc[m][0][r] + sBias1[j];
        const float gf = acc[m][1][r] + sBias1[16 + j];
        const float gz = acc[m][2][r] + sBias1[32 + j];
        const float go = acc[m][3][r] + sBias1[48 + j];
        const float cn = sigm(gf) * c1[m][r] + sigm(gi) * tanhf(gz);
        c1[m][r] = cn;
        const float hn = sigm(go) * tanhf(cn);
        h1n[row * Hh + cblk + j] = tobf(hn);
        prow[r] = hn * sWoutS[j];
      }
      // reduce the 16-column partial dot over lanes sharing the same rows
#pragma unroll
      for (int r = 0; r < 8; ++r) {
        float p = prow[r];
        p += __shfl_xor(p, 1, 32);
        p += __shfl_xor(p, 2, 32);
        p += __shfl_xor(p, 4, 32);
        p += __shfl_xor(p, 8, 32);
        if (j == 0) {
          const int row = mbase + m * 16 + rgrp + r;
          atomicAdd(&preds[t * Bb + row], p);   // preds pre-seeded with b_out
        }
      }
    }
    gbar(cnt, phase);   // preds[t] and h1n visible device-wide
  }
}

// ---------------------------------------------------------------------------
__global__ void loss_kernel(const float* __restrict__ seq,
                            const float* __restrict__ preds, float* out) {
  __shared__ float red[256];
  const int tid = threadIdx.x;
  float s = 0.f;
  for (int i = blockIdx.x * blockDim.x + tid; i < Bb * Tt;
       i += gridDim.x * blockDim.x) {
    const int b = i >> 9;           // i / T
    const int t = i & (Tt - 1);     // i % T
    const float d = seq[i] - preds[t * Bb + b];
    s += d * d;
  }
  red[tid] = s;
  __syncthreads();
  for (int o = 128; o > 0; o >>= 1) {
    if (tid < o) red[tid] += red[tid + o];
    __syncthreads();
  }
  if (tid == 0) atomicAdd(out, red[0] * (1.0f / (float)(Bb * Tt)));
}

// ---------------------------------------------------------------------------
extern "C" void kernel_launch(void* const* d_in, const int* in_sizes, int n_in,
                              void* d_out, int out_size, void* d_ws,
                              size_t ws_size, hipStream_t stream) {
  const float* seq  = (const float*)d_in[0];
  const float* z    = (const float*)d_in[1];
  /* d_in[2] lengths: all == T, unused */
  const float* Wih0 = (const float*)d_in[3];
  const float* Whh0 = (const float*)d_in[4];
  const float* bih0 = (const float*)d_in[5];
  const float* bhh0 = (const float*)d_in[6];
  const float* Wih1 = (const float*)d_in[7];
  const float* Whh1 = (const float*)d_in[8];
  const float* bih1 = (const float*)d_in[9];
  const float* bhh1 = (const float*)d_in[10];
  const float* Wout = (const float*)d_in[11];
  const float* bout = (const float*)d_in[12];
  float* out = (float*)d_out;

  char* ws = (char*)d_ws;
  unsigned* cnt = (unsigned*)ws;                               // 256 B reserved
  float* preds  = (float*)(ws + 256);                          // T*B f32
  bf16*  hbuf   = (bf16*)(ws + 256 + (size_t)Tt * Bb * 4);     // 2 layers x 2 slots x B*H bf16

  init_kernel<<<256, 256, 0, stream>>>(z, bout, preds, hbuf, cnt, out);
  lstm_persistent<<<NWG, 256, 0, stream>>>(z, Wih0, Whh0, bih0, bhh0,
                                           Wih1, Whh1, bih1, bhh1, Wout,
                                           preds, hbuf, cnt);
  loss_kernel<<<128, 256, 0, stream>>>(seq, preds, out);
}
